// LSTMTextEncoder_7473243095591
// MI455X (gfx1250) — matrix operand order repaired and checked by reading, not compiled, not run on hardware
//
#include <hip/hip_runtime.h>

// ---------------------------------------------------------------------------
// LSTM text encoder for MI455X (gfx1250), wave32 + WMMA bf16.
//
//   B=512, T=128, E=H=256, VOCAB=32000, PAD=0.
//   out[b] = normalize( h_T[b] @ fc_w + fc_b ),  h from 128-step LSTM.
//
// Design:
//   * Pack kernel: convert W_g/U_g/fc_w (f32) -> bf16 in the exact CDNA5 WMMA
//     B-operand lane layout (32x16 tile: lane n<16 holds K=0-7,16-23, lane n+16
//     holds K=8-15,24-31). Combined bias bW_g+bU_g in f32.
//   * Main kernel: 32 persistent workgroups, each owns 16 batch rows for the
//     entire T=128 recurrence. h tile (16x256 bf16) lives in LDS; c state in
//     f32 accumulator registers. Per step, x-side (emb gather @ W) and h-side
//     (h @ U) accumulate into the same f32 WMMA accumulators.
//   * Weight base addresses are laundered through empty inline asm each step
//     (prevents LICM hoisting 2KB/lane of loop-invariant fragments -> scratch
//     spills). Loads use GVS addressing: laundered SGPR64 base + 32-bit
//     per-lane byte offset + kt*1024 immediate -> global_load_b128 saddr
//     form, no per-tile 64-bit VALU address math, no WMMA hazard NOPs.
//   * x tile is double-buffered in LDS: step t stages x_{t+1} while WMMAs run.
//   * The reference's pad-mask is forward-identity (m*c+(1-m)*stop_grad(c)==c),
//     so it is skipped. Final FC + row L2-norm fused at the end.
// ---------------------------------------------------------------------------

typedef __attribute__((ext_vector_type(16))) __bf16 v16bf;
typedef __attribute__((ext_vector_type(8)))  float  v8f;
typedef __attribute__((ext_vector_type(4)))  unsigned int u32x4;

// Global (address_space 1) pointers for clean global_load_b128 codegen.
typedef __attribute__((address_space(1))) const u32x4* g4p;
typedef __attribute__((address_space(1))) const char*  gcp;

#define BATCH   512
#define TLEN    128
#define EDIM    256
#define HDIM    256
#define NGATE   1024            // 4 gates * HDIM (order: i, f, o, c)
#define KTILES  8               // 256 / 32
#define PACK_W  (EDIM * NGATE)  // 262144 bf16 elems
#define PACK_FC (HDIM * EDIM)   // 65536

__device__ __forceinline__ float sigf(float x) {
  return 1.0f / (1.0f + __expf(-x));
}

__device__ __forceinline__ v8f splat8(float x) {
  v8f v;
#pragma unroll
  for (int r = 0; r < 8; ++r) v[r] = x;
  return v;
}

// ---------------------------------------------------------------------------
// Weight repack: f32 row-major [K][N] -> bf16 WMMA-B tile layout.
//   tile = e/512, lane = (e/16)%32, slot = e%16
//   ntile = tile/KTILES, kt = tile%KTILES
//   n = ntile*16 + (lane&15)
//   k = kt*32 + ((lane<16)?0:8) + (slot&7) + ((slot&8)?16:0)
// In the hot loop a lane reads 32 contiguous bytes per B fragment.
// ---------------------------------------------------------------------------
__global__ void pack_kernel(
    const float* __restrict__ Wi, const float* __restrict__ Wf,
    const float* __restrict__ Wo, const float* __restrict__ Wc,
    const float* __restrict__ Ui, const float* __restrict__ Uf,
    const float* __restrict__ Uo, const float* __restrict__ Uc,
    const float* __restrict__ bWi, const float* __restrict__ bWf,
    const float* __restrict__ bWo, const float* __restrict__ bWc,
    const float* __restrict__ bUi, const float* __restrict__ bUf,
    const float* __restrict__ bUo, const float* __restrict__ bUc,
    const float* __restrict__ fcw,
    __bf16* __restrict__ Wp, __bf16* __restrict__ Up,
    __bf16* __restrict__ FCp, float* __restrict__ biasC)
{
  const int idx = blockIdx.x * blockDim.x + threadIdx.x;
  if (idx < 2 * PACK_W) {
    const bool isU = idx >= PACK_W;
    const int e = isU ? idx - PACK_W : idx;
    const int tile = e >> 9;
    const int lane = (e >> 4) & 31;
    const int slot = e & 15;
    const int ntile = tile >> 3;     // / KTILES
    const int kt    = tile & 7;
    const int n = ntile * 16 + (lane & 15);
    const int k = kt * 32 + ((lane < 16) ? 0 : 8) + (slot & 7) + ((slot & 8) ? 16 : 0);
    const int g = n >> 8;            // gate 0..3
    const int col = n & 255;
    const float* src;
    if (!isU) src = (g == 0) ? Wi : (g == 1) ? Wf : (g == 2) ? Wo : Wc;
    else      src = (g == 0) ? Ui : (g == 1) ? Uf : (g == 2) ? Uo : Uc;
    const float v = src[k * HDIM + col];
    if (!isU) Wp[e] = (__bf16)v; else Up[e] = (__bf16)v;
  } else if (idx < 2 * PACK_W + PACK_FC) {
    const int e = idx - 2 * PACK_W;
    const int tile = e >> 9;
    const int lane = (e >> 4) & 31;
    const int slot = e & 15;
    const int ntile = tile >> 3;
    const int kt    = tile & 7;
    const int n = ntile * 16 + (lane & 15);
    const int k = kt * 32 + ((lane < 16) ? 0 : 8) + (slot & 7) + ((slot & 8) ? 16 : 0);
    FCp[e] = (__bf16)fcw[k * EDIM + n];
  } else if (idx < 2 * PACK_W + PACK_FC + NGATE) {
    const int j = idx - (2 * PACK_W + PACK_FC);
    const int g = j >> 8;
    const int col = j & 255;
    const float* bw = (g == 0) ? bWi : (g == 1) ? bWf : (g == 2) ? bWo : bWc;
    const float* bu = (g == 0) ? bUi : (g == 1) ? bUf : (g == 2) ? bUo : bUc;
    biasC[j] = bw[col] + bu[col];
  }
}

// ---------------------------------------------------------------------------
// Main fused kernel: 32 blocks x 512 threads (16 waves of 32).
// Block owns batch rows [16*blockIdx, +16). Wave wv owns hidden cols
// [16*wv, +16) of all 4 gates (accumulators share (m,n) so gate math stays
// in-register).
// ---------------------------------------------------------------------------
__launch_bounds__(512, 1)
__global__ void lstm_kernel(
    const int*   __restrict__ captions,   // [B,T]
    const float* __restrict__ emb,        // [VOCAB,E]
    const __bf16* __restrict__ Wp,        // packed [256][1024]
    const __bf16* __restrict__ Up,        // packed [256][1024]
    const __bf16* __restrict__ FCp,       // packed [256][256]
    const float* __restrict__ biasC,      // [4][256] (bW+bU)
    const float* __restrict__ fcb,        // [256]
    float* __restrict__ out)              // [B,E]
{
  __shared__ __align__(16) __bf16 lds_x[2][16 * 256]; // double-buffered x tile
  __shared__ __align__(16) __bf16 lds_h[16 * 256];    // h tile (bf16)
  __shared__ int   lds_tok[16 * TLEN];                // token ids
  __shared__ __align__(16) float lds_o[16 * 256];     // fc output (f32)
  __shared__ float lds_p[16 * 32];                    // norm partials
  __shared__ float lds_inv[16];                       // 1/norm per row

  const int tid  = threadIdx.x;
  const int lane = tid & 31;
  const int wv   = tid >> 5;           // 0..15
  const int b0   = blockIdx.x * 16;

  // Preload all tokens for this block's 16 rows.
  for (int i = tid; i < 16 * TLEN; i += 512) {
    const int r = i >> 7;
    const int t = i & (TLEN - 1);
    lds_tok[i] = captions[(b0 + r) * TLEN + t];
  }
  // h0 = 0
  for (int i = tid; i < 16 * 256; i += 512) lds_h[i] = (__bf16)0.0f;

  // Fragment geometry (ISA layouts):
  //   A (16x32 bf16): lane m<16 holds K 0-7 & 16-23; lane m+16 holds 8-15 & 24-31
  //   C/D (16x16 f32): reg r, lane l -> m = r + (l<16?0:8), n = l&15
  const int mrow  = lane & 15;
  const int moff  = (lane < 16) ? 0 : 8;
  const int abase = (lane < 16) ? 0 : 8;
  const int jcol  = wv * 16 + (lane & 15);    // hidden column of this lane

  const float bias_i = biasC[0 * 256 + jcol];
  const float bias_f = biasC[1 * 256 + jcol];
  const float bias_o = biasC[2 * 256 + jcol];
  const float bias_g = biasC[3 * 256 + jcol];

  v8f cst = splat8(0.0f);              // cell state (rows r+moff, col jcol)

  // Staging assignment: thread (wv,lane) stages row=wv, cols lane*8..+8.
  const int srow = wv;
  const int scol = lane * 8;

  // Per-gate packed-weight 32-bit BYTE offsets (VGPR offset in GVS mode).
  int bofsB[4];
#pragma unroll
  for (int g = 0; g < 4; ++g)
    bofsB[g] = ((((g * 16 + wv) * KTILES) * 512) + lane * 16) * 2;

  __syncthreads();    // tokens + h0 visible

  // Stage x_0 into buffer 0.
  {
    const int tok = lds_tok[srow * TLEN + 0];
    const float* rp = emb + (size_t)tok * EDIM + scol;
    const float4 f0 = *(const float4*)(rp);
    const float4 f1 = *(const float4*)(rp + 4);
    __bf16* dst = &lds_x[0][srow * 256 + scol];
    dst[0] = (__bf16)f0.x; dst[1] = (__bf16)f0.y;
    dst[2] = (__bf16)f0.z; dst[3] = (__bf16)f0.w;
    dst[4] = (__bf16)f1.x; dst[5] = (__bf16)f1.y;
    dst[6] = (__bf16)f1.z; dst[7] = (__bf16)f1.w;
  }
  __syncthreads();

  // Launderable weight base addresses (prevents LICM + scratch spilling).
  unsigned long long wpa = (unsigned long long)Wp;
  unsigned long long upa = (unsigned long long)Up;

  for (int t = 0; t < TLEN; ++t) {
    // Make the weight bases loop-variant: loads must stay inside the loop
    // and stream from L2 each step (weights are L2-resident).
    asm volatile("" : "+s"(wpa), "+s"(upa));
    const gcp wpc = (gcp)wpa;          // SGPR64 base (global AS)
    const gcp upc = (gcp)upa;
    const __bf16* xb = lds_x[t & 1];

    // ---- pre-activations: acc = bias + x@W[:,slice] + h@U[:,slice] ----
    v8f acc0 = splat8(bias_i);
    v8f acc1 = splat8(bias_f);
    v8f acc2 = splat8(bias_o);
    v8f acc3 = splat8(bias_g);

#pragma unroll
    for (int kt = 0; kt < KTILES; ++kt) {
      const int ko = kt * 32 + abase;
      v16bf ax, ah;
      *(u32x4*)&ax       = *(const u32x4*)(xb + mrow * 256 + ko);
      *((u32x4*)&ax + 1) = *(const u32x4*)(xb + mrow * 256 + ko + 16);
      *(u32x4*)&ah       = *(const u32x4*)(lds_h + mrow * 256 + ko);
      *((u32x4*)&ah + 1) = *(const u32x4*)(lds_h + mrow * 256 + ko + 16);

      // Batch all 8 B-fragment loads (16x global_load_b128, GVS addressing:
      // SGPR base + VGPR byte offset + kt*1024 immediate).
      v16bf bw[4], bu[4];
#pragma unroll
      for (int g = 0; g < 4; ++g) {
        const g4p pw = (g4p)(wpc + bofsB[g] + kt * 1024);
        const g4p pu = (g4p)(upc + bofsB[g] + kt * 1024);
        *(u32x4*)&bw[g] = pw[0]; *((u32x4*)&bw[g] + 1) = pw[1];
        *(u32x4*)&bu[g] = pu[0]; *((u32x4*)&bu[g] + 1) = pu[1];
      }
      acc0 = __builtin_amdgcn_wmma_f32_16x16x32_bf16(false, ax, false, bw[0], (short)0, acc0, false, false);
      acc0 = __builtin_amdgcn_wmma_f32_16x16x32_bf16(false, ah, false, bu[0], (short)0, acc0, false, false);
      acc1 = __builtin_amdgcn_wmma_f32_16x16x32_bf16(false, ax, false, bw[1], (short)0, acc1, false, false);
      acc1 = __builtin_amdgcn_wmma_f32_16x16x32_bf16(false, ah, false, bu[1], (short)0, acc1, false, false);
      acc2 = __builtin_amdgcn_wmma_f32_16x16x32_bf16(false, ax, false, bw[2], (short)0, acc2, false, false);
      acc2 = __builtin_amdgcn_wmma_f32_16x16x32_bf16(false, ah, false, bu[2], (short)0, acc2, false, false);
      acc3 = __builtin_amdgcn_wmma_f32_16x16x32_bf16(false, ax, false, bw[3], (short)0, acc3, false, false);
      acc3 = __builtin_amdgcn_wmma_f32_16x16x32_bf16(false, ah, false, bu[3], (short)0, acc3, false, false);
    }

    // ---- stage x_{t+1} into the other buffer while WMMAs drain ----
    if (t + 1 < TLEN) {
      const int tok = lds_tok[srow * TLEN + t + 1];
      const float* rp = emb + (size_t)tok * EDIM + scol;
      const float4 f0 = *(const float4*)(rp);
      const float4 f1 = *(const float4*)(rp + 4);
      __bf16* dst = &lds_x[(t + 1) & 1][srow * 256 + scol];
      dst[0] = (__bf16)f0.x; dst[1] = (__bf16)f0.y;
      dst[2] = (__bf16)f0.z; dst[3] = (__bf16)f0.w;
      dst[4] = (__bf16)f1.x; dst[5] = (__bf16)f1.y;
      dst[6] = (__bf16)f1.z; dst[7] = (__bf16)f1.w;
    }
    __syncthreads();   // (A) all lds_h / lds_x reads of this step are done

    // ---- gate nonlinearities + state update (mask is forward-identity) ----
#pragma unroll
    for (int r = 0; r < 8; ++r) {
      const float iv = sigf(acc0[r]);
      const float fv = sigf(acc1[r]);
      const float ov = sigf(acc2[r]);
      const float gv = tanhf(acc3[r]);
      const float cv = iv * gv + fv * cst[r];
      cst[r] = cv;
      const float hv = ov * tanhf(cv);
      lds_h[(r + moff) * 256 + jcol] = (__bf16)hv;
    }
    __syncthreads();   // (B) h_{t+1} and x_{t+1} visible for next step
  }

  // ---- final FC: out = h @ fc_w + fc_b ; wave wv -> ntile wv of N=256 ----
  v8f acco = splat8(fcb[jcol]);
#pragma unroll
  for (int kt = 0; kt < KTILES; ++kt) {
    const int ko = kt * 32 + abase;
    v16bf ah;
    *(u32x4*)&ah       = *(const u32x4*)(lds_h + mrow * 256 + ko);
    *((u32x4*)&ah + 1) = *(const u32x4*)(lds_h + mrow * 256 + ko + 16);
    const int tileIdx = wv * KTILES + kt;
    const u32x4* pf = (const u32x4*)(FCp + (size_t)tileIdx * 512 + lane * 16);
    v16bf bfr;
    *(u32x4*)&bfr = pf[0]; *((u32x4*)&bfr + 1) = pf[1];
    acco = __builtin_amdgcn_wmma_f32_16x16x32_bf16(
        false, ah, false, bfr, (short)0, acco, false, false);
  }
#pragma unroll
  for (int r = 0; r < 8; ++r) lds_o[(r + moff) * 256 + jcol] = acco[r];
  __syncthreads();

  // ---- row L2 normalization ----
  {
    const int row = tid >> 5;
    const int sub = tid & 31;
    const float* rp = lds_o + row * 256 + sub * 8;
    float s = 0.0f;
#pragma unroll
    for (int q = 0; q < 8; ++q) s += rp[q] * rp[q];
    lds_p[row * 32 + sub] = s;
  }
  __syncthreads();
  if (tid < 16) {
    float s = 0.0f;
    for (int q = 0; q < 32; ++q) s += lds_p[tid * 32 + q];
    float nrm = sqrtf(s);
    nrm = fmaxf(nrm, 1e-12f);
    lds_inv[tid] = 1.0f / nrm;
  }
  __syncthreads();
  {
    const int row = tid >> 5;
    const int sub = tid & 31;
    const float inv = lds_inv[row];
    const float* rp = lds_o + row * 256 + sub * 8;
    float* op = out + (size_t)(b0 + row) * EDIM + sub * 8;
    float4 o0, o1;
    o0.x = rp[0] * inv; o0.y = rp[1] * inv; o0.z = rp[2] * inv; o0.w = rp[3] * inv;
    o1.x = rp[4] * inv; o1.y = rp[5] * inv; o1.z = rp[6] * inv; o1.w = rp[7] * inv;
    *(float4*)op       = o0;
    *(float4*)(op + 4) = o1;
  }
}

// ---------------------------------------------------------------------------
extern "C" void kernel_launch(void* const* d_in, const int* in_sizes, int n_in,
                              void* d_out, int out_size, void* d_ws, size_t ws_size,
                              hipStream_t stream) {
  (void)in_sizes; (void)n_in; (void)out_size; (void)ws_size;

  const int*   captions = (const int*)d_in[0];
  const float* emb      = (const float*)d_in[1];
  const float* Wi = (const float*)d_in[2],  *bWi = (const float*)d_in[3];
  const float* Ui = (const float*)d_in[4],  *bUi = (const float*)d_in[5];
  const float* Wf = (const float*)d_in[6],  *bWf = (const float*)d_in[7];
  const float* Uf = (const float*)d_in[8],  *bUf = (const float*)d_in[9];
  const float* Wo = (const float*)d_in[10], *bWo = (const float*)d_in[11];
  const float* Uo = (const float*)d_in[12], *bUo = (const float*)d_in[13];
  const float* Wc = (const float*)d_in[14], *bWc = (const float*)d_in[15];
  const float* Uc = (const float*)d_in[16], *bUc = (const float*)d_in[17];
  const float* fcw = (const float*)d_in[18];
  const float* fcb = (const float*)d_in[19];

  // Workspace layout (bytes): Wp 512K | Up 512K | FCp 128K | biasC 4K
  char* ws = (char*)d_ws;
  __bf16* Wp    = (__bf16*)(ws);
  __bf16* Up    = (__bf16*)(ws + 524288);
  __bf16* FCp   = (__bf16*)(ws + 1048576);
  float*  biasC = (float*)(ws + 1048576 + 131072);

  const int total = 2 * PACK_W + PACK_FC + NGATE;   // 590848
  pack_kernel<<<(total + 255) / 256, 256, 0, stream>>>(
      Wi, Wf, Wo, Wc, Ui, Uf, Uo, Uc,
      bWi, bWf, bWo, bWc, bUi, bUf, bUo, bUc,
      fcw, Wp, Up, FCp, biasC);

  lstm_kernel<<<BATCH / 16, 512, 0, stream>>>(
      captions, emb, Wp, Up, FCp, biasC, fcb, (float*)d_out);
}
